// HybridSO3FrameDenoisingLayer_20117626814826
// MI455X (gfx1250) — compile-verified, matrix operand order checked by reading, not run on hardware
//
#include <hip/hip_runtime.h>
#include <math.h>

// ---------------------------------------------------------------------------
// HybridSO3FrameDenoisingLayer forward for MI455X (gfx1250, wave32, WMMA).
// Dense matmuls: bf16 WMMA GEMM (v_wmma_f32_16x16x32_bf16, fp32 accumulate)
// with async global->LDS (ASYNCcnt) double-buffered tile pipeline.
// Scatter/segment ops: f32 atomics. Compile-only target.
// ---------------------------------------------------------------------------

#define NN   8192
#define CS   384
#define CZ   128
#define HH   12
#define CH   16
#define HEQ  96
#define FEATS 2112   // H*(C_H + C_Z + P_V*4) = 12*176

typedef __attribute__((ext_vector_type(16))) __bf16 v16bf;
typedef __attribute__((ext_vector_type(8)))  float  v8f;

union BFrag { v16bf v; unsigned int u[8]; };

// ----------------------------- device helpers ------------------------------

__device__ __forceinline__ unsigned short f2bf(float f) {
  unsigned int u = __float_as_uint(f);
  return (unsigned short)((u + 0x7fffu + ((u >> 16) & 1u)) >> 16);
}
__device__ __forceinline__ float act_f(float v, int act) {
  if (act == 1) return v > 0.f ? v : 0.f;          // relu
  if (act == 2) return v / (1.f + expf(-v));       // silu
  if (act == 3) return 1.f / (1.f + expf(-v));     // sigmoid
  return v;
}
__device__ __forceinline__ void atomicMaxF(float* addr, float val) {
  unsigned int* a = (unsigned int*)addr;
  unsigned int old = *a;
  for (int it = 0; it < 64; ++it) {
    float f = __uint_as_float(old);
    if (f >= val) break;
    unsigned int assumed = old;
    old = atomicCAS(a, assumed, __float_as_uint(val));
    if (old == assumed) break;
  }
}

// Generic pointers into LDS carry the LDS byte offset in their low 32 bits
// (aperture encoding), which is exactly what the async-to-LDS VDST expects.
__device__ __forceinline__ unsigned int lds_off32(const void* p) {
  return (unsigned int)(unsigned long long)p;
}
// CDNA5 async DMA: 16B global -> LDS per lane, tracked with ASYNCcnt.
__device__ __forceinline__ void async_ld_b128(unsigned int loff, const unsigned short* g) {
  asm volatile("global_load_async_to_lds_b128 %0, %1, off"
               :: "v"(loff), "v"(g) : "memory");
}
__device__ __forceinline__ void wait_async0() {
  asm volatile("s_wait_asynccnt 0x0" ::: "memory");
}

// ------------------------------- GEMM (WMMA) -------------------------------
// C[M x Nc] = act(A[M x K](bf16) @ B[K x Nc](bf16) + bias), fp32 out.
// 256 threads = 8 waves; 64x64 tile; each wave owns 16x32 (two 16x16 WMMA
// accumulators). Fast path: async b128 global->LDS, double-buffered.

#define GP 40   // A tile pitch (halfwords): 16B-aligned rows, 20-bank stride
#define BP 72   // B tile pitch (halfwords): 16B-aligned rows

__global__ __launch_bounds__(256) void k_gemm_bf16(
    const unsigned short* __restrict__ A, const unsigned short* __restrict__ B,
    const float* __restrict__ bias, float* __restrict__ C,
    int M, int Nc, int K, int act)
{
  __shared__ unsigned short sA[2][64 * GP];   // 64 rows x 32 K
  __shared__ unsigned short sB[2][32 * BP];   // 32 K x 64 N (row-major)
  const int mBase = blockIdx.y * 64;
  const int nBase = blockIdx.x * 64;
  const int tid  = threadIdx.x;
  const int wave = tid >> 5;
  const int lane = tid & 31;
  const int hlf  = lane >> 4;
  const int lrow = lane & 15;
  const int wr = (wave & 3) * 16;
  const int wc = (wave >> 2) * 32;
  v8f acc0 = {};
  v8f acc1 = {};

  const bool fast = (mBase + 64 <= M) && (nBase + 64 <= Nc) &&
                    ((K & 31) == 0) && ((Nc & 7) == 0) && (K >= 32);

  if (fast) {
    const int arow = tid >> 2, acol = (tid & 3) << 3;   // A: 64 rows x 4 chunks
    const int brow = tid >> 3, bcol = (tid & 7) << 3;   // B: 32 rows x 8 chunks
    const unsigned short* gA = A + (size_t)(mBase + arow) * K + acol;
    const unsigned short* gB = B + (size_t)brow * Nc + (nBase + bcol);
    const unsigned int la0 = lds_off32(&sA[0][arow * GP + acol]);
    const unsigned int la1 = lds_off32(&sA[1][arow * GP + acol]);
    const unsigned int lb0 = lds_off32(&sB[0][brow * BP + bcol]);
    const unsigned int lb1 = lds_off32(&sB[1][brow * BP + bcol]);
    async_ld_b128(la0, gA);
    async_ld_b128(lb0, gB);
    const int steps = K >> 5;
    for (int s = 0; s < steps; ++s) {
      wait_async0();        // own tile-s writes landed
      __syncthreads();      // everyone's tile s ready; everyone done reading s-1
      if (s + 1 < steps) {
        const int k1 = (s + 1) << 5;
        async_ld_b128(((s + 1) & 1) ? la1 : la0, gA + k1);
        async_ld_b128(((s + 1) & 1) ? lb1 : lb0, gB + (size_t)k1 * Nc);
      }
      const int buf = s & 1;
      BFrag a, b0, b1;
#pragma unroll
      for (int v = 0; v < 8; ++v) {
        const int koff = (hlf << 3) + ((v < 4) ? (v << 1) : (16 + ((v - 4) << 1)));
        a.u[v] = *(const unsigned int*)&sA[buf][(wr + lrow) * GP + koff];
        unsigned int lo = sB[buf][koff * BP + wc + lrow];
        unsigned int hi = sB[buf][(koff + 1) * BP + wc + lrow];
        b0.u[v] = lo | (hi << 16);
        lo = sB[buf][koff * BP + wc + 16 + lrow];
        hi = sB[buf][(koff + 1) * BP + wc + 16 + lrow];
        b1.u[v] = lo | (hi << 16);
      }
      acc0 = __builtin_amdgcn_wmma_f32_16x16x32_bf16(false, a.v, false, b0.v,
                                                     (short)0, acc0, false, false);
      acc1 = __builtin_amdgcn_wmma_f32_16x16x32_bf16(false, a.v, false, b1.v,
                                                     (short)0, acc1, false, false);
    }
  } else {
    // boundary tiles: guarded scalar fill, single buffer
    for (int k0 = 0; k0 < K; k0 += 32) {
      for (int i = tid; i < 64 * 32; i += 256) {
        int r = i >> 5, c = i & 31;
        int gr = mBase + r, gc = k0 + c;
        sA[0][r * GP + c] = (gr < M && gc < K) ? A[(size_t)gr * K + gc] : (unsigned short)0;
      }
      for (int i = tid; i < 32 * 64; i += 256) {
        int kr = i >> 6, nc = i & 63;
        int gk = k0 + kr, gn = nBase + nc;
        sB[0][kr * BP + nc] = (gk < K && gn < Nc) ? B[(size_t)gk * Nc + gn] : (unsigned short)0;
      }
      if (k0 + 32 < K && tid < 64) {
        int gr = mBase + tid;
        if (gr < M) __builtin_prefetch(&A[(size_t)gr * K + k0 + 32], 0, 1);
      }
      __syncthreads();
      BFrag a, b0, b1;
#pragma unroll
      for (int v = 0; v < 8; ++v) {
        const int koff = (hlf << 3) + ((v < 4) ? (v << 1) : (16 + ((v - 4) << 1)));
        a.u[v] = *(const unsigned int*)&sA[0][(wr + lrow) * GP + koff];
        unsigned int lo = sB[0][koff * BP + wc + lrow];
        unsigned int hi = sB[0][(koff + 1) * BP + wc + lrow];
        b0.u[v] = lo | (hi << 16);
        lo = sB[0][koff * BP + wc + 16 + lrow];
        hi = sB[0][(koff + 1) * BP + wc + 16 + lrow];
        b1.u[v] = lo | (hi << 16);
      }
      acc0 = __builtin_amdgcn_wmma_f32_16x16x32_bf16(false, a.v, false, b0.v,
                                                     (short)0, acc0, false, false);
      acc1 = __builtin_amdgcn_wmma_f32_16x16x32_bf16(false, a.v, false, b1.v,
                                                     (short)0, acc1, false, false);
      __syncthreads();
    }
  }

#pragma unroll
  for (int r = 0; r < 8; ++r) {
    int row = mBase + wr + hlf * 8 + r;
    if (row >= M) continue;
    int col0 = nBase + wc + lrow;
    int col1 = col0 + 16;
    if (col0 < Nc) {
      float v = acc0[r] + (bias ? bias[col0] : 0.f);
      C[(size_t)row * Nc + col0] = act_f(v, act);
    }
    if (col1 < Nc) {
      float v = acc1[r] + (bias ? bias[col1] : 0.f);
      C[(size_t)row * Nc + col1] = act_f(v, act);
    }
  }
}

// --------------------------- small utility kernels -------------------------

__global__ void k_f32_to_bf16(const float* __restrict__ x, unsigned short* __restrict__ y, long long n) {
  for (long long i = blockIdx.x * (long long)blockDim.x + threadIdx.x; i < n;
       i += (long long)gridDim.x * blockDim.x) y[i] = f2bf(x[i]);
}
__global__ void k_fill(float* __restrict__ x, float v, long long n) {
  for (long long i = blockIdx.x * (long long)blockDim.x + threadIdx.x; i < n;
       i += (long long)gridDim.x * blockDim.x) x[i] = v;
}

__global__ void k_quat_rot_mask(const float* __restrict__ quats, const unsigned char* __restrict__ xmask,
                                float* __restrict__ rot, float* __restrict__ mvec, int n) {
  int i = blockIdx.x * blockDim.x + threadIdx.x;
  if (i >= n) return;
  float w = quats[i * 4 + 0], x = quats[i * 4 + 1], y = quats[i * 4 + 2], z = quats[i * 4 + 3];
  float nrm = sqrtf(w * w + x * x + y * y + z * z) + 1e-8f;
  w /= nrm; x /= nrm; y /= nrm; z /= nrm;
  float* R = rot + (size_t)i * 9;
  R[0] = 1.f - 2.f * (y * y + z * z); R[1] = 2.f * (x * y - w * z); R[2] = 2.f * (x * z + w * y);
  R[3] = 2.f * (x * y + w * z); R[4] = 1.f - 2.f * (x * x + z * z); R[5] = 2.f * (y * z - w * x);
  R[6] = 2.f * (x * z - w * y); R[7] = 2.f * (y * z + w * x); R[8] = 1.f - 2.f * (x * x + y * y);
  mvec[i] = xmask[i] ? 0.f : 1.f;
}

// pts (n, npts, 3) in-place: world = rot @ local + trans
__global__ void k_point_frame(float* __restrict__ pts, const float* __restrict__ rot,
                              const float* __restrict__ trans, int n, int npts) {
  long long total = (long long)n * npts;
  for (long long i = blockIdx.x * (long long)blockDim.x + threadIdx.x; i < total;
       i += (long long)gridDim.x * blockDim.x) {
    int node = (int)(i / npts), p = (int)(i % npts);
    float* q = pts + (size_t)node * npts * 3 + (size_t)p * 3;
    const float* R = rot + (size_t)node * 9;
    float a = q[0], b = q[1], c = q[2];
    q[0] = R[0] * a + R[1] * b + R[2] * c + trans[node * 3 + 0];
    q[1] = R[3] * a + R[4] * b + R[5] * c + trans[node * 3 + 1];
    q[2] = R[6] * a + R[7] * b + R[8] * c + trans[node * 3 + 2];
  }
}

__global__ void k_ipa_logits(const float* __restrict__ q, const float* __restrict__ kv,
                             const float* __restrict__ qp, const float* __restrict__ kvp,
                             const float* __restrict__ bmat, const float* __restrict__ gamma_p,
                             const float* __restrict__ mvec, const int* __restrict__ src,
                             const int* __restrict__ dst, float* __restrict__ a, int nE) {
  long long total = (long long)nE * HH;
  for (long long i = blockIdx.x * (long long)blockDim.x + threadIdx.x; i < total;
       i += (long long)gridDim.x * blockDim.x) {
    int e = (int)(i / HH), h = (int)(i % HH);
    int s = src[e], d = dst[e];
    float dot = 0.f;
#pragma unroll
    for (int c = 0; c < CH; ++c)
      dot += q[(size_t)d * (HH * CH) + h * CH + c] * kv[(size_t)s * (2 * HH * CH) + h * 32 + c];
    float sd2 = 0.f;
#pragma unroll
    for (int p = 0; p < 4; ++p) {
#pragma unroll
      for (int dd = 0; dd < 3; ++dd) {
        float df = qp[(size_t)d * 144 + (h * 4 + p) * 3 + dd] -
                   kvp[(size_t)s * 432 + (h * 12 + p) * 3 + dd];
        sd2 += df * df;
      }
    }
    float g = gamma_p[h];
    g = (g > 20.f) ? g : log1pf(expf(g));   // softplus
    float val = dot * 0.14433756729740643f
              + bmat[(size_t)e * HH + h] * 0.5773502691896258f
              - 0.06804138174397717f * g * sd2
              + (mvec[s] - 1.f) * 100000.f;
    a[i] = val;
  }
}

__global__ void k_seg_max(const float* __restrict__ a, const int* __restrict__ dst,
                          float* __restrict__ maxb, int nE) {
  long long total = (long long)nE * HH;
  for (long long i = blockIdx.x * (long long)blockDim.x + threadIdx.x; i < total;
       i += (long long)gridDim.x * blockDim.x) {
    int e = (int)(i / HH), h = (int)(i % HH);
    atomicMaxF(&maxb[(size_t)dst[e] * HH + h], a[i]);
  }
}
__global__ void k_seg_fixmax(float* __restrict__ maxb, long long n) {
  for (long long i = blockIdx.x * (long long)blockDim.x + threadIdx.x; i < n;
       i += (long long)gridDim.x * blockDim.x)
    if (maxb[i] < -1e37f) maxb[i] = 0.f;
}
__global__ void k_seg_expsum(float* __restrict__ a, const int* __restrict__ dst,
                             const float* __restrict__ maxb, float* __restrict__ den, int nE) {
  long long total = (long long)nE * HH;
  for (long long i = blockIdx.x * (long long)blockDim.x + threadIdx.x; i < total;
       i += (long long)gridDim.x * blockDim.x) {
    int e = (int)(i / HH), h = (int)(i % HH);
    float ev = expf(a[i] - maxb[(size_t)dst[e] * HH + h]);
    a[i] = ev;
    atomicAdd(&den[(size_t)dst[e] * HH + h], ev);
  }
}
__global__ void k_seg_norm(float* __restrict__ a, const int* __restrict__ dst,
                           const float* __restrict__ den, int nE) {
  long long total = (long long)nE * HH;
  for (long long i = blockIdx.x * (long long)blockDim.x + threadIdx.x; i < total;
       i += (long long)gridDim.x * blockDim.x) {
    int e = (int)(i / HH), h = (int)(i % HH);
    a[i] = a[i] / (den[(size_t)dst[e] * HH + h] + 1e-8f);
  }
}

__global__ void k_agg_o(const float* __restrict__ attn, const float* __restrict__ kv,
                        const int* __restrict__ src, const int* __restrict__ dst,
                        float* __restrict__ feats, int nE) {
  long long total = (long long)nE * (HH * CH);
  for (long long i = blockIdx.x * (long long)blockDim.x + threadIdx.x; i < total;
       i += (long long)gridDim.x * blockDim.x) {
    int e = (int)(i / (HH * CH)), hc = (int)(i % (HH * CH));
    int h = hc >> 4, c = hc & 15;
    atomicAdd(&feats[(size_t)dst[e] * FEATS + hc],
              attn[(size_t)e * HH + h] * kv[(size_t)src[e] * 768 + h * 32 + 16 + c]);
  }
}
__global__ void k_agg_opt(const float* __restrict__ attn, const float* __restrict__ kvp,
                          const int* __restrict__ src, const int* __restrict__ dst,
                          float* __restrict__ optacc, int nE) {
  long long total = (long long)nE * 288;
  for (long long i = blockIdx.x * (long long)blockDim.x + threadIdx.x; i < total;
       i += (long long)gridDim.x * blockDim.x) {
    int e = (int)(i / 288), t = (int)(i % 288);
    int h = t / 24, rem = t % 24, p = rem / 3, dd = rem % 3;
    atomicAdd(&optacc[(size_t)dst[e] * 288 + t],
              attn[(size_t)e * HH + h] * kvp[(size_t)src[e] * 432 + (h * 12 + 4 + p) * 3 + dd]);
  }
}
__global__ void k_agg_oz(const float* __restrict__ attn, const float* __restrict__ z,
                         const int* __restrict__ dst, float* __restrict__ feats, int nE) {
  long long total = (long long)nE * CZ;
  for (long long i = blockIdx.x * (long long)blockDim.x + threadIdx.x; i < total;
       i += (long long)gridDim.x * blockDim.x) {
    int e = (int)(i / CZ), c = (int)(i % CZ);
    float zv = z[(size_t)e * CZ + c];
    float* base = &feats[(size_t)dst[e] * FEATS + 576 + c];
#pragma unroll
    for (int h = 0; h < HH; ++h)
      atomicAdd(base + h * CZ, attn[(size_t)e * HH + h] * zv);
  }
}
__global__ void k_opt_finalize(const float* __restrict__ optacc, const float* __restrict__ rot,
                               const float* __restrict__ trans, float* __restrict__ feats, int n) {
  long long total = (long long)n * 96;
  for (long long i = blockIdx.x * (long long)blockDim.x + threadIdx.x; i < total;
       i += (long long)gridDim.x * blockDim.x) {
    int node = (int)(i / 96), t = (int)(i % 96);
    const float* R = rot + (size_t)node * 9;
    float a = optacc[(size_t)node * 288 + t * 3 + 0] - trans[node * 3 + 0];
    float b = optacc[(size_t)node * 288 + t * 3 + 1] - trans[node * 3 + 1];
    float c = optacc[(size_t)node * 288 + t * 3 + 2] - trans[node * 3 + 2];
    float o0 = R[0] * a + R[3] * b + R[6] * c;   // rot^T
    float o1 = R[1] * a + R[4] * b + R[7] * c;
    float o2 = R[2] * a + R[5] * b + R[8] * c;
    float* f = feats + (size_t)node * FEATS;
    f[192 + t * 3 + 0] = o0; f[192 + t * 3 + 1] = o1; f[192 + t * 3 + 2] = o2;
    f[480 + t] = sqrtf(o0 * o0 + o1 * o1 + o2 * o2 + 1e-8f);
  }
}

// LayerNorm: out = LN((base? base:0) + upd*(mvec? mvec[row]:1)) * g + b
__global__ void k_ln(const float* __restrict__ base, const float* __restrict__ upd,
                     const float* __restrict__ mvec, const float* __restrict__ g,
                     const float* __restrict__ b, float* __restrict__ out, int n, int D) {
  int row = blockIdx.x;
  if (row >= n) return;
  __shared__ float red[128];
  float vloc[4];
  int cnt = 0;
  float s = 0.f;
  for (int c = threadIdx.x; c < D; c += 128) {
    float v = upd[(size_t)row * D + c];
    if (mvec) v *= mvec[row];
    if (base) v += base[(size_t)row * D + c];
    vloc[cnt++] = v;
    s += v;
  }
  red[threadIdx.x] = s; __syncthreads();
  for (int st = 64; st > 0; st >>= 1) {
    if (threadIdx.x < st) red[threadIdx.x] += red[threadIdx.x + st];
    __syncthreads();
  }
  float mean = red[0] / (float)D;
  __syncthreads();
  float ss = 0.f; cnt = 0;
  for (int c = threadIdx.x; c < D; c += 128) { float d2 = vloc[cnt++] - mean; ss += d2 * d2; }
  red[threadIdx.x] = ss; __syncthreads();
  for (int st = 64; st > 0; st >>= 1) {
    if (threadIdx.x < st) red[threadIdx.x] += red[threadIdx.x + st];
    __syncthreads();
  }
  float inv = rsqrtf(red[0] / (float)D + 1e-5f);
  cnt = 0;
  for (int c = threadIdx.x; c < D; c += 128)
    out[(size_t)row * D + c] = (vloc[cnt++] - mean) * inv * g[c] + b[c];
}

__global__ void k_eq_norm(const float* __restrict__ x, const float* __restrict__ g0,
                          const float* __restrict__ b0, const float* __restrict__ g1,
                          float* __restrict__ h, int n) {
  int row = blockIdx.x;
  if (row >= n) return;
  __shared__ float r0[128], r1[128], r2[128];
  int c = threadIdx.x;
  float sc = 0.f, vsq = 0.f, v1 = 0.f, v2 = 0.f, v3 = 0.f;
  if (c < HEQ) {
    const float* xr = x + (size_t)row * 384;
    sc = xr[c]; v1 = xr[96 + c]; v2 = xr[192 + c]; v3 = xr[288 + c];
    vsq = v1 * v1 + v2 * v2 + v3 * v3;
  }
  r0[c] = sc; r1[c] = sc * sc; r2[c] = vsq; __syncthreads();
  for (int st = 64; st > 0; st >>= 1) {
    if (c < st) { r0[c] += r0[c + st]; r1[c] += r1[c + st]; r2[c] += r2[c + st]; }
    __syncthreads();
  }
  float mean = r0[0] / 96.f;
  float var = r1[0] / 96.f - mean * mean;
  float vn = sqrtf(r2[0] / 96.f + 1e-8f);
  if (c < HEQ) {
    float* hr = h + (size_t)row * 384;
    hr[c] = (sc - mean) * rsqrtf(var + 1e-5f) * g0[c] + b0[c];
    float gc = g1[c] / vn;
    hr[96 + c] = v1 * gc; hr[192 + c] = v2 * gc; hr[288 + c] = v3 * gc;
  }
}

__global__ void k_edge_wigner(const float* __restrict__ trans, const int* __restrict__ src,
                              const int* __restrict__ dst, float* __restrict__ Dw, int nE) {
  for (long long e = blockIdx.x * (long long)blockDim.x + threadIdx.x; e < nE;
       e += (long long)gridDim.x * blockDim.x) {
    int s = src[e], d = dst[e];
    float vx = trans[d * 3 + 0] - trans[s * 3 + 0];
    float vy = trans[d * 3 + 1] - trans[s * 3 + 1];
    float vz = trans[d * 3 + 2] - trans[s * 3 + 2];
    float nrm = sqrtf(vx * vx + vy * vy + vz * vz) + 1e-8f;
    float dx = vx / nrm, dy = vy / nrm, dz = vz / nrm;
    float ux, uy, uz;
    if (fabsf(dz) < 0.99f) { ux = 0.f; uy = 0.f; uz = 1.f; }
    else                   { ux = 1.f; uy = 0.f; uz = 0.f; }
    float xa0 = uy * dz - uz * dy, xa1 = uz * dx - ux * dz, xa2 = ux * dy - uy * dx;
    float xn = sqrtf(xa0 * xa0 + xa1 * xa1 + xa2 * xa2) + 1e-8f;
    xa0 /= xn; xa1 /= xn; xa2 /= xn;
    float za0 = dy * xa2 - dz * xa1, za1 = dz * xa0 - dx * xa2, za2 = dx * xa1 - dy * xa0;
    float R[9] = { xa0, xa1, xa2, dx, dy, dz, za0, za1, za2 };   // rows: xa, d, za
    float* D1 = Dw + (size_t)e * 9;
#pragma unroll
    for (int i = 0; i < 3; ++i)
#pragma unroll
      for (int j = 0; j < 3; ++j)
        D1[i * 3 + j] = R[((i + 1) % 3) * 3 + ((j + 1) % 3)];
  }
}

__global__ void k_cat_s2in(const float* __restrict__ so3, const float* __restrict__ node,
                           unsigned short* __restrict__ out, int n) {
  long long total = (long long)n * 480;
  for (long long i = blockIdx.x * (long long)blockDim.x + threadIdx.x; i < total;
       i += (long long)gridDim.x * blockDim.x) {
    int r = (int)(i / 480), c = (int)(i % 480);
    float v = (c < 96) ? so3[(size_t)r * 384 + c] : node[(size_t)r * 384 + (c - 96)];
    out[i] = f2bf(v);
  }
}
__global__ void k_set_so3row0(const float* __restrict__ so3, const float* __restrict__ s2,
                              float* __restrict__ x, int n) {
  long long total = (long long)n * 384;
  for (long long i = blockIdx.x * (long long)blockDim.x + threadIdx.x; i < total;
       i += (long long)gridDim.x * blockDim.x) {
    int r = (int)(i / 384), c = (int)(i % 384);
    x[i] = (c < 96) ? s2[(size_t)r * 96 + c] : so3[i];
  }
}
__global__ void k_apply_D(const float* __restrict__ h, const float* __restrict__ Dw,
                          const int* __restrict__ src, unsigned short* __restrict__ xs, int nE) {
  long long total = (long long)nE * 96;
  for (long long i = blockIdx.x * (long long)blockDim.x + threadIdx.x; i < total;
       i += (long long)gridDim.x * blockDim.x) {
    int e = (int)(i / 96), c = (int)(i % 96);
    const float* hr = h + (size_t)src[e] * 384;
    const float* D1 = Dw + (size_t)e * 9;
    float h1 = hr[96 + c], h2 = hr[192 + c], h3 = hr[288 + c];
    unsigned short* xr = xs + (size_t)e * 384;
    xr[c] = f2bf(hr[c]);
    xr[96 + c]  = f2bf(D1[0] * h1 + D1[1] * h2 + D1[2] * h3);
    xr[192 + c] = f2bf(D1[3] * h1 + D1[4] * h2 + D1[5] * h3);
    xr[288 + c] = f2bf(D1[6] * h1 + D1[7] * h2 + D1[8] * h3);
  }
}
__global__ void k_cat_invin(const float* __restrict__ h, const float* __restrict__ ez,
                            const int* __restrict__ src, const int* __restrict__ dst,
                            unsigned short* __restrict__ out, int nE) {
  long long total = (long long)nE * 320;
  for (long long i = blockIdx.x * (long long)blockDim.x + threadIdx.x; i < total;
       i += (long long)gridDim.x * blockDim.x) {
    int e = (int)(i / 320), c = (int)(i % 320);
    float v;
    if (c < 96)       v = h[(size_t)src[e] * 384 + c];
    else if (c < 192) v = h[(size_t)dst[e] * 384 + (c - 96)];
    else              v = ez[(size_t)e * 128 + (c - 192)];
    out[i] = f2bf(v);
  }
}
__global__ void k_msg_transform(const float* __restrict__ v0, const float* __restrict__ v1all,
                                const float* __restrict__ gate, const float* __restrict__ alpha,
                                const float* __restrict__ Dw, const int* __restrict__ dst,
                                float* __restrict__ agg, int nE) {
  long long total = (long long)nE * 96;
  for (long long i = blockIdx.x * (long long)blockDim.x + threadIdx.x; i < total;
       i += (long long)gridDim.x * blockDim.x) {
    int e = (int)(i / 96), c = (int)(i % 96);
    int d = dst[e];
    float aw = alpha[(size_t)e * HH + (c >> 3)];
    float gt = gate[(size_t)e * 96 + c] * aw;
    float m0 = v0[(size_t)e * 96 + c] * aw;
    const float* vr = v1all + (size_t)e * 384;
    float m1 = vr[96 + c] * gt, m2 = vr[192 + c] * gt, m3 = vr[288 + c] * gt;
    const float* D1 = Dw + (size_t)e * 9;
    float o1 = D1[0] * m1 + D1[3] * m2 + D1[6] * m3;   // D^T
    float o2 = D1[1] * m1 + D1[4] * m2 + D1[7] * m3;
    float o3 = D1[2] * m1 + D1[5] * m2 + D1[8] * m3;
    atomicAdd(&agg[(size_t)d * 384 + c], m0);
    atomicAdd(&agg[(size_t)d * 384 + 96 + c], o1);
    atomicAdd(&agg[(size_t)d * 384 + 192 + c], o2);
    atomicAdd(&agg[(size_t)d * 384 + 288 + c], o3);
  }
}
__global__ void k_gather_row0_bf16(const float* __restrict__ x, unsigned short* __restrict__ y,
                                   int n, int rowstride, int cols) {
  long long total = (long long)n * cols;
  for (long long i = blockIdx.x * (long long)blockDim.x + threadIdx.x; i < total;
       i += (long long)gridDim.x * blockDim.x) {
    int r = (int)(i / cols), c = (int)(i % cols);
    y[i] = f2bf(x[(size_t)r * rowstride + c]);
  }
}
__global__ void k_so3_combine1(const float* __restrict__ x0, const float* __restrict__ upd0,
                               const float* __restrict__ updall, float* __restrict__ x1, int n) {
  long long total = (long long)n * 384;
  for (long long i = blockIdx.x * (long long)blockDim.x + threadIdx.x; i < total;
       i += (long long)gridDim.x * blockDim.x) {
    int r = (int)(i / 384), c = (int)(i % 384);
    float u = (c < 96) ? upd0[(size_t)r * 96 + c] : updall[i];
    x1[i] = x0[i] + u;
  }
}
__global__ void k_so3_combine2(const float* __restrict__ x1, const float* __restrict__ s_out,
                               const float* __restrict__ voutall, const float* __restrict__ g2,
                               float* __restrict__ out, int n) {
  long long total = (long long)n * 384;
  for (long long i = blockIdx.x * (long long)blockDim.x + threadIdx.x; i < total;
       i += (long long)gridDim.x * blockDim.x) {
    int r = (int)(i / 384), c = (int)(i % 384);
    float u = (c < 96) ? s_out[(size_t)r * 96 + c]
                       : voutall[i] * g2[(size_t)r * 96 + (c % 96)];
    out[i] = x1[i] + u;
  }
}
__global__ void k_mask_mul(const float* __restrict__ x, const float* __restrict__ mvec,
                           float* __restrict__ y, int n, int D) {
  long long total = (long long)n * D;
  for (long long i = blockIdx.x * (long long)blockDim.x + threadIdx.x; i < total;
       i += (long long)gridDim.x * blockDim.x)
    y[i] = x[i] * mvec[i / D];
}
__global__ void k_bb(const float* __restrict__ node, const float* __restrict__ nm,
                     const float* __restrict__ w, const float* __restrict__ b,
                     float* __restrict__ bb, int n) {
  long long total = (long long)n * 6;
  for (long long i = blockIdx.x * (long long)blockDim.x + threadIdx.x; i < total;
       i += (long long)gridDim.x * blockDim.x) {
    int r = (int)(i / 6), j = (int)(i % 6);
    float m = nm[r];
    float s = 0.f;
    for (int c = 0; c < CS; ++c) s += node[(size_t)r * CS + c] * w[c * 6 + j];
    bb[i] = (s * m + b[j]) * m;
  }
}
__global__ void k_compose(const float* __restrict__ rot, const float* __restrict__ trans,
                          const float* __restrict__ bb, float* __restrict__ rotO,
                          float* __restrict__ transO, int n) {
  int i = blockIdx.x * blockDim.x + threadIdx.x;
  if (i >= n) return;
  const float* R = rot + (size_t)i * 9;
  float b = bb[i * 6], c = bb[i * 6 + 1], d = bb[i * 6 + 2];
  float t0 = bb[i * 6 + 3], t1 = bb[i * 6 + 4], t2 = bb[i * 6 + 5];
  float nrm = sqrtf(1.f + b * b + c * c + d * d) + 1e-8f;
  float w = 1.f / nrm; b /= nrm; c /= nrm; d /= nrm;
  float Ru[9] = { 1.f - 2.f * (c * c + d * d), 2.f * (b * c - w * d), 2.f * (b * d + w * c),
                  2.f * (b * c + w * d), 1.f - 2.f * (b * b + d * d), 2.f * (c * d - w * b),
                  2.f * (b * d - w * c), 2.f * (c * d + w * b), 1.f - 2.f * (b * b + c * c) };
  float* Ro = rotO + (size_t)i * 9;
#pragma unroll
  for (int ii = 0; ii < 3; ++ii)
#pragma unroll
    for (int jj = 0; jj < 3; ++jj)
      Ro[ii * 3 + jj] = R[ii * 3 + 0] * Ru[jj] + R[ii * 3 + 1] * Ru[3 + jj] + R[ii * 3 + 2] * Ru[6 + jj];
#pragma unroll
  for (int ii = 0; ii < 3; ++ii)
    transO[i * 3 + ii] = trans[i * 3 + ii] + R[ii * 3 + 0] * t0 + R[ii * 3 + 1] * t1 + R[ii * 3 + 2] * t2;
}
__global__ void k_cat_et(const unsigned short* __restrict__ hdown, const float* __restrict__ z,
                         const int* __restrict__ src, const int* __restrict__ dst,
                         unsigned short* __restrict__ out, int nE) {
  long long total = (long long)nE * 512;
  for (long long i = blockIdx.x * (long long)blockDim.x + threadIdx.x; i < total;
       i += (long long)gridDim.x * blockDim.x) {
    int e = (int)(i >> 9), c = (int)(i & 511);
    unsigned short v;
    if (c < 192)      v = hdown[(size_t)src[e] * 192 + c];
    else if (c < 384) v = hdown[(size_t)dst[e] * 192 + (c - 192)];
    else              v = f2bf(z[(size_t)e * 128 + (c - 384)]);
    out[i] = v;
  }
}

// ------------------------------- host side ---------------------------------

struct Arena { char* base; size_t off; };
template <typename T> static T* aalloc(Arena& ar, size_t count) {
  size_t bytes = ((count * sizeof(T)) + 255) & ~(size_t)255;
  T* p = (T*)(ar.base + ar.off);
  ar.off += bytes;
  return p;
}
static inline int nblk(long long n, int t) {
  long long b = (n + t - 1) / t;
  if (b > 1048576) b = 1048576;
  if (b < 1) b = 1;
  return (int)b;
}

struct IpaW { unsigned short *wq, *wkv, *wqp, *wkvp, *wb, *wo; size_t bo, gamma; };

extern "C" void kernel_launch(void* const* d_in, const int* in_sizes, int n_in,
                              void* d_out, int out_size, void* d_ws, size_t ws_size,
                              hipStream_t stream) {
  (void)in_sizes; (void)n_in; (void)out_size; (void)ws_size;
  const int E  = 245760;
  const int ES = 131072;

  const float* in_node = (const float*)d_in[0];
  const float* in_quat = (const float*)d_in[1];
  const float* in_trans = (const float*)d_in[2];
  const float* in_z  = (const float*)d_in[3];
  const float* in_zs = (const float*)d_in[4];
  const float* in_so3 = (const float*)d_in[5];
  const float* in_nm = (const float*)d_in[6];
  const float* P = (const float*)d_in[7];
  const int* eidx  = (const int*)d_in[8];
  const int* esidx = (const int*)d_in[9];
  const unsigned char* in_xmask = (const unsigned char*)d_in[10];

  // -------- parameter offsets (pytree sorted-key flatten order) ----------
  size_t off = 0;
  auto T = [&](size_t n) { size_t o = off; off += n; return o; };
  size_t p_bb_b = T(6), p_bb_w = T((size_t)CS * 6);
  size_t p_et_b1 = T(512), p_et_b2 = T(128), p_et_down_b = T(192), p_et_down_w = T((size_t)CS * 192),
         p_et_ln_b = T(128), p_et_ln_g = T(128), p_et_w1 = T(512 * 512), p_et_w2 = T(512 * 128);
  struct IpaOff { size_t bo, gamma, wb, wkv, wkvp, wo, wq, wqp; };
  auto takeIpa = [&]() {
    IpaOff q;
    q.bo = T(CS); q.gamma = T(HH); q.wb = T((size_t)CZ * HH); q.wkv = T((size_t)CS * 384);
    q.wkvp = T((size_t)CS * 432); q.wo = T((size_t)FEATS * CS); q.wq = T((size_t)CS * 192);
    q.wqp = T((size_t)CS * 144);
    return q;
  };
  IpaOff ipaSeqO = takeIpa();
  IpaOff ipaSpO  = takeIpa();
  size_t p_ln1_b = T(CS), p_ln1_g = T(CS), p_ln2_b = T(CS), p_ln2_g = T(CS),
         p_ln3_b = T(CS), p_ln3_g = T(CS);
  size_t p_nt_b1 = T(CS), p_nt_b2 = T(CS), p_nt_b3 = T(CS), p_nt_lnb = T(CS), p_nt_lng = T(CS),
         p_nt_w1 = T((size_t)CS * CS), p_nt_w2 = T((size_t)CS * CS), p_nt_w3 = T((size_t)CS * CS);
  size_t p_set_b1 = T(512), p_set_b2 = T(128), p_set_down_b = T(192), p_set_down_w = T((size_t)CS * 192),
         p_set_ln_b = T(128), p_set_ln_g = T(128), p_set_w1 = T(512 * 512), p_set_w2 = T(512 * 128);
  size_t p_aw = T(96 * 12), p_eb1 = T(128), p_eb2 = T(128), p_ew1 = T(128 * 128), p_ew2 = T(128 * 128),
         p_fb1 = T(192), p_fb2 = T(96), p_fw1 = T(96 * 192), p_fw2 = T(192 * 96), p_fwg = T(192 * 96),
         p_fwv = T(96 * 96), p_gw = T(96 * 96), p_mb = T(96), p_mw = T(320 * 96),
         p_n1b = T(96), p_n1g = T(96), p_n1v = T(96), p_n2b = T(96), p_n2g = T(96), p_n2v = T(96),
         p_o0 = T(96 * 96), p_o1 = T(96 * 96), p_v0 = T(96 * 96), p_v1 = T(96 * 96);
  size_t p_tc_b = T(96), p_tc_w = T(480 * 96), p_te_b = T(CS), p_te_w = T((size_t)96 * CS);

  // ------------------------ output sections ------------------------------
  float* out = (float*)d_out;
  float* out_node  = out;
  float* out_rot   = out_node + (size_t)NN * CS;
  float* out_trans = out_rot + (size_t)NN * 9;
  float* out_z     = out_trans + (size_t)NN * 3;
  float* out_zs    = out_z + (size_t)E * CZ;
  float* out_so3   = out_zs + (size_t)ES * CZ;

  Arena ar{ (char*)d_ws, 0 };

  auto conv = [&](const float* x, unsigned short* y, long long n) {
    k_f32_to_bf16<<<nblk(n, 256), 256, 0, stream>>>(x, y, n);
  };
  auto fill = [&](float* x, float v, long long n) {
    k_fill<<<nblk(n, 256), 256, 0, stream>>>(x, v, n);
  };
  auto gemm = [&](const unsigned short* A, const unsigned short* B, const float* bias,
                  float* C, int M, int Nc, int K, int act) {
    dim3 g((Nc + 63) / 64, (M + 63) / 64);
    k_gemm_bf16<<<g, 256, 0, stream>>>(A, B, bias, C, M, Nc, K, act);
  };
  auto cvtW = [&](size_t poff, size_t n) -> unsigned short* {
    unsigned short* w = aalloc<unsigned short>(ar, n);
    conv(P + poff, w, (long long)n);
    return w;
  };

  // ---------------------- persistent allocations -------------------------
  float* rotbuf = aalloc<float>(ar, (size_t)NN * 9);
  float* mvec   = aalloc<float>(ar, NN);
  float* nodeA  = aalloc<float>(ar, (size_t)NN * CS);
  float* nodeB  = aalloc<float>(ar, (size_t)NN * CS);
  float* updbuf = aalloc<float>(ar, (size_t)NN * CS);
  unsigned short* node_bf = aalloc<unsigned short>(ar, (size_t)NN * CS);
  unsigned short* zbf  = aalloc<unsigned short>(ar, (size_t)E * CZ);
  unsigned short* zsbf = aalloc<unsigned short>(ar, (size_t)ES * CZ);

  // weights -> bf16 (persistent)
  IpaW WSp, WSeq;
  auto loadIpaW = [&](const IpaOff& o) {
    IpaW w;
    w.wq = cvtW(o.wq, (size_t)CS * 192); w.wkv = cvtW(o.wkv, (size_t)CS * 384);
    w.wqp = cvtW(o.wqp, (size_t)CS * 144); w.wkvp = cvtW(o.wkvp, (size_t)CS * 432);
    w.wb = cvtW(o.wb, (size_t)CZ * HH); w.wo = cvtW(o.wo, (size_t)FEATS * CS);
    w.bo = o.bo; w.gamma = o.gamma;
    return w;
  };
  WSeq = loadIpaW(ipaSeqO);
  WSp  = loadIpaW(ipaSpO);
  unsigned short* w_tc = cvtW(p_tc_w, 480 * 96);
  unsigned short* w_te = cvtW(p_te_w, (size_t)96 * CS);
  unsigned short* w_ew1 = cvtW(p_ew1, 128 * 128);
  unsigned short* w_ew2 = cvtW(p_ew2, 128 * 128);
  unsigned short* w_mw = cvtW(p_mw, 320 * 96);
  unsigned short* w_aw = cvtW(p_aw, 96 * 12);
  unsigned short* w_v0 = cvtW(p_v0, 96 * 96);
  unsigned short* w_v1 = cvtW(p_v1, 96 * 96);
  unsigned short* w_gw = cvtW(p_gw, 96 * 96);
  unsigned short* w_o0 = cvtW(p_o0, 96 * 96);
  unsigned short* w_o1 = cvtW(p_o1, 96 * 96);
  unsigned short* w_fw1 = cvtW(p_fw1, 96 * 192);
  unsigned short* w_fw2 = cvtW(p_fw2, 192 * 96);
  unsigned short* w_fwg = cvtW(p_fwg, 192 * 96);
  unsigned short* w_fwv = cvtW(p_fwv, 96 * 96);
  unsigned short* w_nt1 = cvtW(p_nt_w1, (size_t)CS * CS);
  unsigned short* w_nt2 = cvtW(p_nt_w2, (size_t)CS * CS);
  unsigned short* w_nt3 = cvtW(p_nt_w3, (size_t)CS * CS);
  unsigned short* w_et_down = cvtW(p_et_down_w, (size_t)CS * 192);
  unsigned short* w_et_1 = cvtW(p_et_w1, 512 * 512);
  unsigned short* w_et_2 = cvtW(p_et_w2, 512 * 128);
  unsigned short* w_set_down = cvtW(p_set_down_w, (size_t)CS * 192);
  unsigned short* w_set_1 = cvtW(p_set_w1, 512 * 512);
  unsigned short* w_set_2 = cvtW(p_set_w2, 512 * 128);

  size_t mark = ar.off;   // phase scratch starts here

  // ------------------------ initial conversions --------------------------
  k_quat_rot_mask<<<nblk(NN, 256), 256, 0, stream>>>(in_quat, in_xmask, rotbuf, mvec, NN);
  conv(in_z, zbf, (long long)E * CZ);
  conv(in_zs, zsbf, (long long)ES * CZ);
  conv(in_node, node_bf, (long long)NN * CS);

  // ------------------------------ graph IPA ------------------------------
  auto run_ipa = [&](const unsigned short* s_bf, const IpaW& W, const float* z,
                     const unsigned short* z_bf, const int* idx, int nE, float* upd) {
    ar.off = mark;
    const int* src = idx;
    const int* dst = idx + nE;
    float* q   = aalloc<float>(ar, (size_t)NN * 192);
    float* kv  = aalloc<float>(ar, (size_t)NN * 384);
    float* qp  = aalloc<float>(ar, (size_t)NN * 144);
    float* kvp = aalloc<float>(ar, (size_t)NN * 432);
    gemm(s_bf, W.wq, nullptr, q, NN, 192, CS, 0);
    gemm(s_bf, W.wkv, nullptr, kv, NN, 384, CS, 0);
    gemm(s_bf, W.wqp, nullptr, qp, NN, 144, CS, 0);
    gemm(s_bf, W.wkvp, nullptr, kvp, NN, 432, CS, 0);
    k_point_frame<<<nblk((long long)NN * 48, 256), 256, 0, stream>>>(qp, rotbuf, in_trans, NN, 48);
    k_point_frame<<<nblk((long long)NN * 144, 256), 256, 0, stream>>>(kvp, rotbuf, in_trans, NN, 144);
    float* bmat = aalloc<float>(ar, (size_t)nE * HH);
    gemm(z_bf, W.wb, nullptr, bmat, nE, HH, CZ, 0);
    float* a = aalloc<float>(ar, (size_t)nE * HH);
    k_ipa_logits<<<nblk((long long)nE * HH, 256), 256, 0, stream>>>(
        q, kv, qp, kvp, bmat, P + W.gamma, mvec, src, dst, a, nE);
    float* maxb = aalloc<float>(ar, (size_t)NN * HH);
    float* den  = aalloc<float>(ar, (size_t)NN * HH);
    fill(maxb, -3.0e38f, (long long)NN * HH);
    fill(den, 0.f, (long long)NN * HH);
    k_seg_max<<<nblk((long long)nE * HH, 256), 256, 0, stream>>>(a, dst, maxb, nE);
    k_seg_fixmax<<<nblk((long long)NN * HH, 256), 256, 0, stream>>>(maxb, (long long)NN * HH);
    k_seg_expsum<<<nblk((long long)nE * HH, 256), 256, 0, stream>>>(a, dst, maxb, den, nE);
    k_seg_norm<<<nblk((long long)nE * HH, 256), 256, 0, stream>>>(a, dst, den, nE);
    float* feats  = aalloc<float>(ar, (size_t)NN * FEATS);
    float* optacc = aalloc<float>(ar, (size_t)NN * 288);
    fill(feats, 0.f, (long long)NN * FEATS);
    fill(optacc, 0.f, (long long)NN * 288);
    k_agg_o<<<nblk((long long)nE * HH * CH, 256), 256, 0, stream>>>(a, kv, src, dst, feats, nE);
    k_agg_opt<<<nblk((long long)nE * 288, 256), 256, 0, stream>>>(a, kvp, src, dst, optacc, nE);
    k_agg_oz<<<nblk((long long)nE * CZ, 256), 256, 0, stream>>>(a, z, dst, feats, nE);
    k_opt_finalize<<<nblk((long long)NN * 96, 256), 256, 0, stream>>>(optacc, rotbuf, in_trans, feats, NN);
    unsigned short* feats_bf = aalloc<unsigned short>(ar, (size_t)NN * FEATS);
    conv(feats, feats_bf, (long long)NN * FEATS);
    gemm(feats_bf, W.wo, P + W.bo, upd, NN, CS, FEATS, 0);
  };

  run_ipa(node_bf, WSp, in_z, zbf, eidx, E, updbuf);
  k_ln<<<NN, 128, 0, stream>>>(in_node, updbuf, mvec, P + p_ln1_g, P + p_ln1_b, nodeA, NN, CS);
  conv(nodeA, node_bf, (long long)NN * CS);

  run_ipa(node_bf, WSeq, in_zs, zsbf, esidx, ES, updbuf);
  k_ln<<<NN, 128, 0, stream>>>(nodeA, updbuf, mvec, P + p_ln2_g, P + p_ln2_b, nodeB, NN, CS);
  conv(nodeB, node_bf, (long long)NN * CS);

  // ------------------------------ SO(3) block ----------------------------
  {
    ar.off = mark;
    const int* src = eidx;
    const int* dst = eidx + E;
    float* Dw = aalloc<float>(ar, (size_t)E * 9);
    k_edge_wigner<<<nblk(E, 256), 256, 0, stream>>>(in_trans, src, dst, Dw, E);

    unsigned short* catbf = aalloc<unsigned short>(ar, (size_t)NN * 480);
    k_cat_s2in<<<nblk((long long)NN * 480, 256), 256, 0, stream>>>(in_so3, nodeB, catbf, NN);
    float* s2 = aalloc<float>(ar, (size_t)NN * 96);
    gemm(catbf, w_tc, P + p_tc_b, s2, NN, 96, 480, 0);
    float* so3x = aalloc<float>(ar, (size_t)NN * 384);
    k_set_so3row0<<<nblk((long long)NN * 384, 256), 256, 0, stream>>>(in_so3, s2, so3x, NN);

    float* h = aalloc<float>(ar, (size_t)NN * 384);
    k_eq_norm<<<NN, 128, 0, stream>>>(so3x, P + p_n1g, P + p_n1b, P + p_n1v, h, NN);

    unsigned short* xs_bf = aalloc<unsigned short>(ar, (size_t)E * 384);
    k_apply_D<<<nblk((long long)E * 96, 256), 256, 0, stream>>>(h, Dw, src, xs_bf, E);

    float* ezt = aalloc<float>(ar, (size_t)E * 128);
    gemm(zbf, w_ew1, P + p_eb1, ezt, E, 128, 128, 2);   // silu
    unsigned short* eztbf = aalloc<unsigned short>(ar, (size_t)E * 128);
    conv(ezt, eztbf, (long long)E * 128);
    float* ez = aalloc<float>(ar, (size_t)E * 128);
    gemm(eztbf, w_ew2, P + p_eb2, ez, E, 128, 128, 0);

    unsigned short* invin = aalloc<unsigned short>(ar, (size_t)E * 320);
    k_cat_invin<<<nblk((long long)E * 320, 256), 256, 0, stream>>>(h, ez, src, dst, invin, E);
    float* inv = aalloc<float>(ar, (size_t)E * 96);
    gemm(invin, w_mw, P + p_mb, inv, E, 96, 320, 2);    // silu
    unsigned short* inv_bf = aalloc<unsigned short>(ar, (size_t)E * 96);
    conv(inv, inv_bf, (long long)E * 96);

    float* alpha = aalloc<float>(ar, (size_t)E * HH);
    gemm(inv_bf, w_aw, nullptr, alpha, E, HH, 96, 0);
    float* maxb = aalloc<float>(ar, (size_t)NN * HH);
    float* den  = aalloc<float>(ar, (size_t)NN * HH);
    fill(maxb, -3.0e38f, (long long)NN * HH);
    fill(den, 0.f, (long long)NN * HH);
    k_seg_max<<<nblk((long long)E * HH, 256), 256, 0, stream>>>(alpha, dst, maxb, E);
    k_seg_fixmax<<<nblk((long long)NN * HH, 256), 256, 0, stream>>>(maxb, (long long)NN * HH);
    k_seg_expsum<<<nblk((long long)E * HH, 256), 256, 0, stream>>>(alpha, dst, maxb, den, E);
    k_seg_norm<<<nblk((long long)E * HH, 256), 256, 0, stream>>>(alpha, dst, den, E);

    float* v0 = aalloc<float>(ar, (size_t)E * 96);
    gemm(inv_bf, w_v0, nullptr, v0, E, 96, 96, 0);
    float* gate = aalloc<float>(ar, (size_t)E * 96);
    gemm(inv_bf, w_gw, nullptr, gate, E, 96, 96, 3);    // sigmoid
    float* v1all = aalloc<float>(ar, (size_t)E * 384);
    gemm(xs_bf, w_v1, nullptr, v1all, E * 4, 96, 96, 0);

    float* agg = aalloc<float>(ar, (size_t)NN * 384);
    fill(agg, 0.f, (long long)NN * 384);
    k_msg_transform<<<nblk((long long)E * 96, 256), 256, 0, stream>>>(
        v0, v1all, gate, alpha, Dw, dst, agg, E);

    unsigned short* agg_bf = aalloc<unsigned short>(ar, (size_t)NN * 384);
    conv(agg, agg_bf, (long long)NN * 384);
    unsigned short* agg0_bf = aalloc<unsigned short>(ar, (size_t)NN * 96);
    k_gather_row0_bf16<<<nblk((long long)NN * 96, 256), 256, 0, stream>>>(agg, agg0_bf, NN, 384, 96);
    float* upd0 = aalloc<float>(ar, (size_t)NN * 96);
    gemm(agg0_bf, w_o0, nullptr, upd0, NN, 96, 96, 0);
    float* updall = aalloc<float>(ar, (size_t)NN * 384);
    gemm(agg_bf, w_o1, nullptr, updall, NN * 4, 96, 96, 0);
    float* x1 = aalloc<float>(ar, (size_t)NN * 384);
    k_so3_combine1<<<nblk((long long)NN * 384, 256), 256, 0, stream>>>(so3x, upd0, updall, x1, NN);

    float* h2 = aalloc<float>(ar, (size_t)NN * 384);
    k_eq_norm<<<NN, 128, 0, stream>>>(x1, P + p_n2g, P + p_n2b, P + p_n2v, h2, NN);
    unsigned short* h2_bf = aalloc<unsigned short>(ar, (size_t)NN * 384);
    conv(h2, h2_bf, (long long)NN * 384);
    unsigned short* h20_bf = aalloc<unsigned short>(ar, (size_t)NN * 96);
    k_gather_row0_bf16<<<nblk((long long)NN * 96, 256), 256, 0, stream>>>(h2, h20_bf, NN, 384, 96);

    float* sh = aalloc<float>(ar, (size_t)NN * 192);
    gemm(h20_bf, w_fw1, P + p_fb1, sh, NN, 192, 96, 2); // silu
    unsigned short* sh_bf = aalloc<unsigned short>(ar, (size_t)NN * 192);
    conv(sh, sh_bf, (long long)NN * 192);
    float* s_out = aalloc<float>(ar, (size_t)NN * 96);
    gemm(sh_bf, w_fw2, P + p_fb2, s_out, NN, 96, 192, 0);
    float* g2 = aalloc<float>(ar, (size_t)NN * 96);
    gemm(sh_bf, w_fwg, nullptr, g2, NN, 96, 192, 3);    // sigmoid
    float* voutall = aalloc<float>(ar, (size_t)NN * 384);
    gemm(h2_bf, w_fwv, nullptr, voutall, NN * 4, 96, 96, 0);
    k_so3_combine2<<<nblk((long long)NN * 384, 256), 256, 0, stream>>>(
        x1, s_out, voutall, g2, out_so3, NN);

    unsigned short* so3r0_bf = aalloc<unsigned short>(ar, (size_t)NN * 96);
    k_gather_row0_bf16<<<nblk((long long)NN * 96, 256), 256, 0, stream>>>(out_so3, so3r0_bf, NN, 384, 96);
    gemm(so3r0_bf, w_te, P + p_te_b, updbuf, NN, CS, 96, 0);
  }
  k_ln<<<NN, 128, 0, stream>>>(nodeB, updbuf, mvec, P + p_ln3_g, P + p_ln3_b, nodeA, NN, CS);

  // --------------------------- node transition ---------------------------
  {
    ar.off = mark;
    conv(nodeA, node_bf, (long long)NN * CS);
    float* h1 = aalloc<float>(ar, (size_t)NN * CS);
    unsigned short* h1b = aalloc<unsigned short>(ar, (size_t)NN * CS);
    float* h2 = aalloc<float>(ar, (size_t)NN * CS);
    unsigned short* h2b = aalloc<unsigned short>(ar, (size_t)NN * CS);
    float* h3 = aalloc<float>(ar, (size_t)NN * CS);
    gemm(node_bf, w_nt1, P + p_nt_b1, h1, NN, CS, CS, 1);  // relu
    conv(h1, h1b, (long long)NN * CS);
    gemm(h1b, w_nt2, P + p_nt_b2, h2, NN, CS, CS, 1);      // relu
    conv(h2, h2b, (long long)NN * CS);
    gemm(h2b, w_nt3, P + p_nt_b3, h3, NN, CS, CS, 0);
    k_ln<<<NN, 128, 0, stream>>>(nodeA, h3, nullptr, P + p_nt_lng, P + p_nt_lnb, nodeB, NN, CS);
  }
  k_mask_mul<<<nblk((long long)NN * CS, 256), 256, 0, stream>>>(nodeB, mvec, out_node, NN, CS);
  conv(out_node, node_bf, (long long)NN * CS);

  // ----------------------- backbone update + compose ---------------------
  {
    ar.off = mark;
    float* bb = aalloc<float>(ar, (size_t)NN * 6);
    k_bb<<<nblk((long long)NN * 6, 256), 256, 0, stream>>>(out_node, in_nm, P + p_bb_w, P + p_bb_b, bb, NN);
    k_compose<<<nblk(NN, 256), 256, 0, stream>>>(rotbuf, in_trans, bb, out_rot, out_trans, NN);
  }

  // --------------------------- edge transitions --------------------------
  auto run_et = [&](const unsigned short* wdown, size_t b_down, const unsigned short* w1,
                    size_t b1, const unsigned short* w2, size_t b2, size_t lng, size_t lnb,
                    const float* z, const int* idx, int nE, float* outz) {
    ar.off = mark;
    const int* src = idx;
    const int* dst = idx + nE;
    float* hdown = aalloc<float>(ar, (size_t)NN * 192);
    gemm(node_bf, wdown, P + b_down, hdown, NN, 192, CS, 0);
    unsigned short* hdown_bf = aalloc<unsigned short>(ar, (size_t)NN * 192);
    conv(hdown, hdown_bf, (long long)NN * 192);
    unsigned short* xcat = aalloc<unsigned short>(ar, (size_t)nE * 512);
    k_cat_et<<<nblk((long long)nE * 512, 256), 256, 0, stream>>>(hdown_bf, z, src, dst, xcat, nE);
    float* t1 = aalloc<float>(ar, (size_t)nE * 512);
    gemm(xcat, w1, P + b1, t1, nE, 512, 512, 1);   // relu
    conv(t1, xcat, (long long)nE * 512);           // reuse xcat as bf16 of t1
    float* t2 = aalloc<float>(ar, (size_t)nE * 128);
    gemm(xcat, w2, P + b2, t2, nE, 128, 512, 0);
    k_ln<<<nE, 128, 0, stream>>>(nullptr, t2, nullptr, P + lng, P + lnb, outz, nE, 128);
  };
  run_et(w_et_down, p_et_down_b, w_et_1, p_et_b1, w_et_2, p_et_b2, p_et_ln_g, p_et_ln_b,
         in_z, eidx, E, out_z);
  run_et(w_set_down, p_set_down_b, w_set_1, p_set_b1, w_set_2, p_set_b2, p_set_ln_g, p_set_ln_b,
         in_zs, esidx, ES, out_zs);
}
